// LocalAttention_7524782702880
// MI455X (gfx1250) — matrix operand order repaired and checked by reading, not compiled, not run on hardware
//
#include <hip/hip_runtime.h>
#include <math.h>

typedef __attribute__((ext_vector_type(16))) _Float16 v16h;
typedef __attribute__((ext_vector_type(8)))  _Float16 h8;
typedef __attribute__((ext_vector_type(4)))  _Float16 h4;
typedef __attribute__((ext_vector_type(8)))  float    v8f;

namespace {
constexpr int   kN = 2, kL = 8192, kH = 16, kE = 64, kS = 8192, kD = 64;
constexpr int   kCtx  = 128, kHalf = 64;
// Fold log2(e) into the softmax temperature so exp() becomes a single v_exp_f32
// (exp2). softmax(temp*x) == 2^(temp*log2e*x - max)/sum, identical result.
constexpr float kTempL      = 0.125f * 1.44269504088896340736f;
constexpr float kMaskScaled = -1.0e9f * 0.125f * 1.44269504088896340736f;

constexpr int kQB    = 32;                   // queries per block = 2 waves x 16
constexpr int kWaves = 2;
constexpr int kRows  = 160;                  // key-band union rows per block
constexpr int kKT    = 9;                    // 144-wide score window per 16-query tile
// LDS row strides (halves): 16B-aligned rows AND 16 consecutive rows hit 16
// distinct banks (stride_dwords * lane mod 64 all distinct for lane=0..15).
constexpr int kKStr  = 72;                   // 36 dwords
constexpr int kVStr  = 184;                  // 92 dwords; cols 0..159 data, 160..175 zero pad
constexpr int kPStr  = 168;                  // 84 dwords; cols 0..143 data, 144..159 zero pad

__device__ inline v8f wmma16(v16h a, v16h b, v8f c) {
  // D = A(16x32 f16) * B(32x16 f16) + C(16x16 f32)
  return __builtin_amdgcn_wmma_f32_16x16x32_f16(false, a, false, b, (short)0, c,
                                                false, false);
}
// 16 contiguous halves as two aligned 16B LDS loads
__device__ inline v16h ld16(const _Float16* p) {
  h8 lo = *(const h8*)p;
  h8 hi = *(const h8*)(p + 8);
  return __builtin_shufflevector(lo, hi, 0, 1, 2, 3, 4, 5, 6, 7, 8, 9, 10, 11, 12, 13, 14, 15);
}
__device__ inline h8 cvt_h8(float4 a, float4 b) {
  h8 r;
  r[0] = (_Float16)a.x; r[1] = (_Float16)a.y; r[2] = (_Float16)a.z; r[3] = (_Float16)a.w;
  r[4] = (_Float16)b.x; r[5] = (_Float16)b.y; r[6] = (_Float16)b.z; r[7] = (_Float16)b.w;
  return r;
}
__device__ inline h4 cvt_h4(float4 a) {
  h4 r;
  r[0] = (_Float16)a.x; r[1] = (_Float16)a.y; r[2] = (_Float16)a.z; r[3] = (_Float16)a.w;
  return r;
}
} // namespace

__global__ __launch_bounds__(kWaves * 32)
void local_attn_wmma(const float* __restrict__ Q, const float* __restrict__ K,
                     const float* __restrict__ V, const float* __restrict__ AM,
                     const int* __restrict__ KLEN, float* __restrict__ OUT) {
  __shared__ __align__(16) _Float16 Klds[kRows * kKStr];        // 23040 B
  __shared__ __align__(16) _Float16 Vtlds[kD * kVStr];          // 23552 B  [d][band col]
  __shared__ __align__(16) _Float16 Plds[kWaves * 16 * kPStr];  // 10752 B

  const int QBLKS = kL / kQB;                 // 256
  const int bid = blockIdx.x;
  const int n   = bid / (kH * QBLKS);
  const int rem = bid % (kH * QBLKS);
  const int h   = rem / QBLKS;
  const int q0  = (rem % QBLKS) * kQB;
  const int jbase = q0 - kHalf;

  const int tid  = threadIdx.x;
  const int wave = tid >> 5;
  const int lane = tid & 31;
  const int m    = lane & 15;
  const int g    = lane >> 4;
  const int klen = KLEN[n];

  // Wave-uniform fast-path predicate: whole band in-range and below key_length.
  const bool interior = (jbase >= 0) && (jbase + kRows <= kS);
  const bool full     = interior && (jbase + kRows <= klen);

  // ---- Stage K band: f32 global -> f16 LDS rows (ds_store_b64) ----
  {
    const size_t kbase = ((size_t)n * kS) * (kH * kE) + (size_t)h * kE;
    const int e4  = (tid & 15) * 4;
    const int jl0 = tid >> 4;                 // 0..3, stride 4
    const float* kp = K + kbase + (size_t)(jbase + jl0) * (kH * kE) + e4;
    _Float16*   kd = &Klds[jl0 * kKStr + e4];
    if (interior) {
      for (int jl = jl0; jl < kRows; jl += 4) {
        *(h4*)kd = cvt_h4(*(const float4*)kp);
        kp += 4 * (kH * kE);
        kd += 4 * kKStr;
      }
    } else {
      for (int jl = jl0; jl < kRows; jl += 4) {
        const int j = jbase + jl;
        float4 val = make_float4(0.f, 0.f, 0.f, 0.f);
        if (j >= 0 && j < kS) val = *(const float4*)kp;
        *(h4*)kd = cvt_h4(val);
        kp += 4 * (kH * kE);
        kd += 4 * kKStr;
      }
    }
  }
  // ---- Stage V band transposed via 4x4 register transpose ----
  {
    const size_t vbase = ((size_t)n * kS) * (kH * kD) + (size_t)h * kD;
    const int d0  = (tid & 15) * 4;
    const int j00 = (tid >> 4) * 4;           // 0,4,8,12; stride 16
    const float* vp = V + vbase + (size_t)(jbase + j00) * (kH * kD) + d0;
    for (int j0 = j00; j0 < kRows; j0 += 16) {
      float4 r0 = make_float4(0.f, 0.f, 0.f, 0.f), r1 = r0, r2 = r0, r3 = r0;
      // j block of 4 is fully in or fully out of [0,kS) (all bounds mult of 4)
      if (interior || ((jbase + j0) >= 0 && (jbase + j0) < kS)) {
        r0 = *(const float4*)vp;
        r1 = *(const float4*)(vp + kH * kD);
        r2 = *(const float4*)(vp + 2 * kH * kD);
        r3 = *(const float4*)(vp + 3 * kH * kD);
      }
      h4 c0, c1, c2, c3;
      c0[0]=(_Float16)r0.x; c0[1]=(_Float16)r1.x; c0[2]=(_Float16)r2.x; c0[3]=(_Float16)r3.x;
      c1[0]=(_Float16)r0.y; c1[1]=(_Float16)r1.y; c1[2]=(_Float16)r2.y; c1[3]=(_Float16)r3.y;
      c2[0]=(_Float16)r0.z; c2[1]=(_Float16)r1.z; c2[2]=(_Float16)r2.z; c2[3]=(_Float16)r3.z;
      c3[0]=(_Float16)r0.w; c3[1]=(_Float16)r1.w; c3[2]=(_Float16)r2.w; c3[3]=(_Float16)r3.w;
      _Float16* vd = &Vtlds[d0 * kVStr + j0];
      *(h4*)(vd)             = c0;
      *(h4*)(vd + kVStr)     = c1;
      *(h4*)(vd + 2 * kVStr) = c2;
      *(h4*)(vd + 3 * kVStr) = c3;
      vp += 16 * (kH * kD);
    }
  }
  // ---- Zero-fill V^T pad cols [160,176) so padded P cols multiply exact zeros ----
  {
    h4 z = {};
    for (int f = tid; f < kD * 4; f += kWaves * 32) {
      *(h4*)&Vtlds[(f >> 2) * kVStr + 160 + (f & 3) * 4] = z;
    }
  }
  __syncthreads();

  const int i0 = q0 + 16 * wave;

  // ---- Q A-fragments: A 16x32 f16; lane row = lane&15; halves 0..7 -> K=8g+h,
  //      halves 8..15 -> K=16+8g+h. Vectorized float4 loads + packed cvt. ----
  const float* qrow = Q + ((size_t)(n * kL + i0 + m) * kH + h) * kE;
  const float4* q4 = (const float4*)qrow;
  const float4 f0 = q4[2 * g],      f1 = q4[2 * g + 1];
  const float4 f2 = q4[4 + 2 * g],  f3 = q4[5 + 2 * g];
  const float4 f4 = q4[8 + 2 * g],  f5 = q4[9 + 2 * g];
  const float4 f6 = q4[12 + 2 * g], f7 = q4[13 + 2 * g];
  const v16h a0 = __builtin_shufflevector(cvt_h8(f0, f1), cvt_h8(f2, f3),
                                          0,1,2,3,4,5,6,7,8,9,10,11,12,13,14,15);
  const v16h a1 = __builtin_shufflevector(cvt_h8(f4, f5), cvt_h8(f6, f7),
                                          0,1,2,3,4,5,6,7,8,9,10,11,12,13,14,15);

  // ---- QK^T over 9 key tiles (B 32x16: lane col = lane&15, halves 0..7 -> K=8g+h, 8..15 -> K=16+8g+h) ----
  v8f acc[kKT];
#pragma unroll
  for (int kt = 0; kt < kKT; ++kt) {
    const _Float16* krow = &Klds[(16 * wave + 16 * kt + m) * kKStr];
    const v16h b0 = __builtin_shufflevector(*(const h8*)(krow + 8 * g), *(const h8*)(krow + 16 + 8 * g), 0,1,2,3,4,5,6,7,8,9,10,11,12,13,14,15);
    const v16h b1 = __builtin_shufflevector(*(const h8*)(krow + 32 + 8 * g), *(const h8*)(krow + 48 + 8 * g), 0,1,2,3,4,5,6,7,8,9,10,11,12,13,14,15);
    v8f c = {};
    c = wmma16(a0, b0, c);
    c = wmma16(a1, b1, c);
    acc[kt] = c;
  }

  // ---- mask + scale + attn_mask (C layout: row = r+8g, col = lane&15) ----
  if (full) {
#pragma unroll
    for (int kt = 0; kt < kKT; ++kt) {
      const int j = jbase + 16 * wave + 16 * kt + m;
      const float* amp = AM + (size_t)(i0 + 8 * g) * kS + j;
#pragma unroll
      for (int r = 0; r < 8; ++r) {
        const float am = amp[(size_t)r * kS];
        const unsigned off = (unsigned)(j - (i0 + r + 8 * g) + kHalf);
        acc[kt][r] = (off < (unsigned)kCtx) ? kTempL * (acc[kt][r] + am) : kMaskScaled;
      }
    }
  } else {
#pragma unroll
    for (int kt = 0; kt < kKT; ++kt) {
      const int j  = jbase + 16 * wave + 16 * kt + m;
      const bool jr = (j >= 0) && (j < kS) && (j < klen);
      const int jc = j < 0 ? 0 : (j > kS - 1 ? kS - 1 : j);
      const float* amp = AM + (size_t)(i0 + 8 * g) * kS + jc;
#pragma unroll
      for (int r = 0; r < 8; ++r) {
        const float am = amp[(size_t)r * kS];
        const unsigned off = (unsigned)(j - (i0 + r + 8 * g) + kHalf);
        const bool valid = jr && (off < (unsigned)kCtx);
        acc[kt][r] = valid ? kTempL * (acc[kt][r] + am) : kMaskScaled;
      }
    }
  }

  // ---- softmax per row across the 16 lanes of each half-group.
  //      P stores unnormalized exp; 1/rsum applied to the final output. ----
  float isv[8];
#pragma unroll
  for (int r = 0; r < 8; ++r) {
    float rmax = -3.0e38f;
#pragma unroll
    for (int kt = 0; kt < kKT; ++kt) rmax = fmaxf(rmax, acc[kt][r]);
#pragma unroll
    for (int off = 1; off < 16; off <<= 1)
      rmax = fmaxf(rmax, __shfl_xor(rmax, off, 32));
    float rsum = 0.f;
#pragma unroll
    for (int kt = 0; kt < kKT; ++kt) {
      const float p = exp2f(acc[kt][r] - rmax);   // single v_exp_f32
      acc[kt][r] = p;
      rsum += p;
    }
#pragma unroll
    for (int off = 1; off < 16; off <<= 1)
      rsum += __shfl_xor(rsum, off, 32);
    isv[r] = __builtin_amdgcn_rcpf(rsum);
    const int row = r + 8 * g;
#pragma unroll
    for (int kt = 0; kt < kKT; ++kt)
      Plds[(wave * 16 + row) * kPStr + 16 * kt + m] = (_Float16)acc[kt][r];
  }
  // zero-pad P band cols [144,160)
  {
    h4 z = {};
    for (int f = lane; f < 16 * 4; f += 32) {
      *(h4*)&Plds[(wave * 16 + (f >> 2)) * kPStr + 144 + (f & 3) * 4] = z;
    }
  }
  __syncthreads();

  // ---- P(16x160) x V(160x64): hoisted P A-fragments, 4 d-tiles x 5 k-steps ----
  const _Float16* prow = &Plds[(wave * 16 + m) * kPStr];
  v16h pa[5];
#pragma unroll
  for (int t = 0; t < 5; ++t) {
    h8 lo = *(const h8*)(prow + 32 * t + 16 * g);
    h8 hi = *(const h8*)(prow + 32 * t + 16 * g + 8);
    pa[t] = __builtin_shufflevector(lo, hi, 0,1,2,3,4,5,6,7,8,9,10,11,12,13,14,15);
  }
#pragma unroll
  for (int dt = 0; dt < 4; ++dt) {
    v8f o = {};
    const int d = 16 * dt + m;
    const _Float16* vrow = &Vtlds[d * kVStr + 16 * wave];
#pragma unroll
    for (int t = 0; t < 5; ++t) {
      const v16h vb = ld16(vrow + 32 * t + 16 * g);
      o = wmma16(pa[t], vb, o);
    }
#pragma unroll
    for (int r = 0; r < 8; ++r) {
      const int i = i0 + r + 8 * g;
      OUT[((size_t)(n * kL + i) * kH + h) * kD + d] = o[r] * isv[r];
    }
  }
}

extern "C" void kernel_launch(void* const* d_in, const int* in_sizes, int n_in,
                              void* d_out, int out_size, void* d_ws, size_t ws_size,
                              hipStream_t stream) {
  (void)in_sizes; (void)n_in; (void)out_size; (void)d_ws; (void)ws_size;
  const float* q    = (const float*)d_in[0];
  const float* k    = (const float*)d_in[1];
  const float* v    = (const float*)d_in[2];
  const float* am   = (const float*)d_in[3];
  const int*   klen = (const int*)d_in[5];   // d_in[4] = query_lengths (unused by reference)
  float* out = (float*)d_out;

  dim3 grid(kN * kH * (kL / kQB));           // 8192 blocks
  dim3 block(kWaves * 32);                   // 64 threads = 2 wave32
  hipLaunchKernelGGL(local_attn_wmma, grid, block, 0, stream, q, k, v, am, klen, out);
}
